// BBGRUDecoder_23708219474710
// MI455X (gfx1250) — matrix operand order repaired and checked by reading, not compiled, not run on hardware
//
#include <hip/hip_runtime.h>
#include <hip/hip_bf16.h>
#include <math.h>

typedef __attribute__((ext_vector_type(16))) _Float16 v16h;
typedef __attribute__((ext_vector_type(8)))  _Float16 v8h;
typedef __attribute__((ext_vector_type(8)))  float    v8f;

#define N_NODES 500000
#define N_EDGES 4000000
#define F_INP   10
#define FDIM    64
#define HDIM    128
#define G3H     384
#define BATCH   2048
#define GMAX    20
#define NGR     (BATCH*GMAX)
#define KH      12
#define DHD     32
#define TSTEPS  GMAX

static __device__ __forceinline__ v16h cat16(v8h a, v8h b) {
    return __builtin_shufflevector(a, b, 0,1,2,3,4,5,6,7,8,9,10,11,12,13,14,15);
}

// ---------- f32 -> f16 convert ----------
__global__ void cvt_f32_f16(const float* __restrict__ in, _Float16* __restrict__ out, int n) {
    int i = blockIdx.x * blockDim.x + threadIdx.x;
    if (i < n) out[i] = (_Float16)in[i];
}

// Build transposed+concatenated conv1 weight: Btc[n][k] = k<64 ? W_root1[k][n] : W_nbr1[k-64][n]
__global__ void build_btc(const float* __restrict__ Wr, const float* __restrict__ Wn,
                          _Float16* __restrict__ btc) {
    int i = blockIdx.x * blockDim.x + threadIdx.x;           // 64*128
    if (i >= FDIM * HDIM) return;
    int n = i / HDIM, k = i % HDIM;
    float v = (k < FDIM) ? Wr[k * FDIM + n] : Wn[(k - FDIM) * FDIM + n];
    btc[i] = (_Float16)v;
}

// ---------- conv0: edge aggregation (f32 atomics, 10 features) ----------
__global__ void edge_agg0(const int* __restrict__ src, const int* __restrict__ dst,
                          const float* __restrict__ ew, const float* __restrict__ x,
                          float* __restrict__ agg0) {
    long long t = (long long)blockIdx.x * blockDim.x + threadIdx.x;
    int f = (int)(t & 15);
    long long e = t >> 4;
    if (e >= N_EDGES || f >= F_INP) return;
    int s = src[e], d = dst[e];
    float v = x[(long long)s * F_INP + f] * ew[e];
    atomicAdd(&agg0[(long long)d * F_INP + f], v);
}

// ---------- conv0: node dense (10->64), writes f16 h1 ----------
__global__ void conv0_node(const float* __restrict__ x, const float* __restrict__ agg0,
                           const float* __restrict__ Wr, const float* __restrict__ Wn,
                           const float* __restrict__ b0, _Float16* __restrict__ h1) {
    long long t = (long long)blockIdx.x * blockDim.x + threadIdx.x;
    if (t >= (long long)N_NODES * FDIM) return;
    int j = (int)(t & (FDIM - 1));
    long long n = t >> 6;
    float acc = b0[j];
    const float* xr = x + n * F_INP;
    const float* ar = agg0 + n * F_INP;
#pragma unroll
    for (int i = 0; i < F_INP; ++i)
        acc = fmaf(xr[i], Wr[i * FDIM + j], fmaf(ar[i], Wn[i * FDIM + j], acc));
    h1[t] = (_Float16)fmaxf(acc, 0.0f);
}

// ---------- conv1: edge aggregation via CDNA5 packed-f16 atomics ----------
// one wave per edge; lane handles one f16 pair (32 pairs = 64 features)
__global__ void edge_agg1(const int* __restrict__ src, const int* __restrict__ dst,
                          const float* __restrict__ ew, const _Float16* __restrict__ h1,
                          _Float16* __restrict__ agg) {
    long long t = (long long)blockIdx.x * blockDim.x + threadIdx.x;
    int lane = (int)(t & 31);
    long long e = t >> 5;
    if (e >= N_EDGES) return;
    int s = src[e], d = dst[e];
    float w = ew[e];
    union { unsigned u; _Float16 h[2]; } v, r;
    v.u = *(const unsigned*)(h1 + (long long)s * FDIM + 2 * lane);
    r.h[0] = (_Float16)((float)v.h[0] * w);
    r.h[1] = (_Float16)((float)v.h[1] * w);
    _Float16* outp = agg + (long long)d * FDIM + 2 * lane;
    asm volatile("global_atomic_pk_add_f16 %0, %1, off"
                 :: "v"(outp), "v"(r.u) : "memory");
}

// ---------- count nodes per graph label ----------
__global__ void count_labels(const int* __restrict__ labels, int* __restrict__ cnts) {
    int n = blockIdx.x * blockDim.x + threadIdx.x;
    if (n < N_NODES) atomicAdd(&cnts[labels[n]], 1);
}

// ---------- conv1 dense (WMMA) fused with mean-pool scatter-add ----------
// wave computes a 16x64 strip (4 accumulators, A loaded once per k-step) of
// relu([h1|agg] @ [Wr1;Wn1] + b1), then atomically accumulates into
// sums[label[node]][col].
__global__ void conv1_pool(const _Float16* __restrict__ h1, const _Float16* __restrict__ agg,
                           const _Float16* __restrict__ btc, const float* __restrict__ b1,
                           const int* __restrict__ labels, float* __restrict__ sums) {
    int wave = (blockIdx.x * blockDim.x + threadIdx.x) >> 5;
    int lane = threadIdx.x & 31;
    const int total = N_NODES / 16;           // 31250 strips
    if (wave >= total) return;
    int tm = wave;
    int arow = tm * 16 + (lane & 15);
    int koffA = (lane < 16) ? 0 : 8;
    int koffB = (lane < 16) ? 0 : 8;
    const _Float16* A0 = h1  + (long long)arow * FDIM;
    const _Float16* A1 = agg + (long long)arow * FDIM;
    const _Float16* Bp[4];
#pragma unroll
    for (int u = 0; u < 4; ++u)
        Bp[u] = btc + (long long)(u * 16 + (lane & 15)) * HDIM + koffB;
    v8f acc[4] = {};
#pragma unroll
    for (int s = 0; s < 4; ++s) {
        const _Float16* Ap = (s < 2) ? A0 : A1;
        int ks = (s & 1) * 32;
        v16h av = cat16(*(const v8h*)(Ap + ks + koffA),
                        *(const v8h*)(Ap + ks + 16 + koffA));
#pragma unroll
        for (int u = 0; u < 4; ++u) {
            v16h bv = cat16(*(const v8h*)(Bp[u] + s * 32),
                            *(const v8h*)(Bp[u] + s * 32 + 16));
            acc[u] = __builtin_amdgcn_wmma_f32_16x16x32_f16(false, av, false, bv,
                                                            (short)0, acc[u], false, false);
        }
    }
    int col0 = lane & 15;
    int mbase = tm * 16 + ((lane < 16) ? 0 : 8);
#pragma unroll
    for (int r = 0; r < 8; ++r) {
        int node = mbase + r;
        long long srow = (long long)labels[node] * FDIM;
#pragma unroll
        for (int u = 0; u < 4; ++u) {
            float v = fmaxf(acc[u][r] + b1[u * 16 + col0], 0.0f);
            atomicAdd(&sums[srow + u * 16 + col0], v);
        }
    }
}

// ---------- fill xs with empty embedding, then scatter pooled embeddings ----------
__global__ void fill_xs(const float* __restrict__ emptyE, _Float16* __restrict__ xs) {
    int i = blockIdx.x * blockDim.x + threadIdx.x;
    if (i < TSTEPS * BATCH * FDIM) xs[i] = (_Float16)emptyE[i & (FDIM - 1)];
}

__global__ void scatter_xs(const float* __restrict__ sums, const int* __restrict__ cnts,
                           const int* __restrict__ label_map, _Float16* __restrict__ xs) {
    int i = blockIdx.x * blockDim.x + threadIdx.x;   // NGR*FDIM
    if (i >= NGR * FDIM) return;
    int g = i >> 6, f = i & (FDIM - 1);
    float c = (float)cnts[g];
    float v = sums[i] / fmaxf(c, 1.0f);
    int lm = label_map[g];
    int b = lm / GMAX, slot = lm % GMAX;
    xs[((long long)slot * BATCH + b) * FDIM + f] = (_Float16)v;
}

// ---------- WMMA GEMM, 16x64 strip per wave:
// C[M,N] = A[M,K](f16,row) @ Bt[N,K](f16,row)^T, N multiple of 64 ----------
__global__ void gemm16x64(const _Float16* __restrict__ A, int lda,
                          const _Float16* __restrict__ Bt, int ldb,
                          float* __restrict__ C, int ldc,
                          int Ng, int total, int K) {
    int wave = (blockIdx.x * blockDim.x + threadIdx.x) >> 5;
    int lane = threadIdx.x & 31;
    if (wave >= total) return;
    int tm = wave / Ng, tg = wave % Ng;
    int arow = tm * 16 + (lane & 15);
    int koffA = (lane < 16) ? 0 : 8;
    int koffB = (lane < 16) ? 0 : 8;
    const _Float16* Ap = A + (long long)arow * lda;
    const _Float16* Bp[4];
#pragma unroll
    for (int u = 0; u < 4; ++u)
        Bp[u] = Bt + (long long)(tg * 64 + u * 16 + (lane & 15)) * ldb + koffB;
    v8f acc[4] = {};
    for (int ks = 0; ks < K; ks += 32) {
        v16h av = cat16(*(const v8h*)(Ap + ks + koffA),
                        *(const v8h*)(Ap + ks + 16 + koffA));
#pragma unroll
        for (int u = 0; u < 4; ++u) {
            v16h bv = cat16(*(const v8h*)(Bp[u] + ks),
                            *(const v8h*)(Bp[u] + ks + 16));
            acc[u] = __builtin_amdgcn_wmma_f32_16x16x32_f16(false, av, false, bv,
                                                            (short)0, acc[u], false, false);
        }
    }
    float* Crow = C + (long long)(tm * 16 + ((lane < 16) ? 0 : 8)) * ldc
                    + tg * 64 + (lane & 15);
#pragma unroll
    for (int u = 0; u < 4; ++u)
#pragma unroll
        for (int r = 0; r < 8; ++r)
            Crow[(long long)r * ldc + u * 16] = acc[u][r];
}

// ---------- GRU gate elementwise (in-place h update) ----------
__global__ void gru_gate(const float* __restrict__ GI, const float* __restrict__ GH,
                         const float* __restrict__ bih, const float* __restrict__ bhh,
                         float* __restrict__ h, _Float16* __restrict__ hh,
                         _Float16* __restrict__ ys) {
    int i = blockIdx.x * blockDim.x + threadIdx.x;   // BATCH*HDIM
    if (i >= BATCH * HDIM) return;
    int b = i >> 7, j = i & (HDIM - 1);
    const float* gi = GI + (long long)b * G3H;
    const float* gh = GH + (long long)b * G3H;
    float ir = gi[j]            + bih[j];
    float iz = gi[HDIM + j]     + bih[HDIM + j];
    float in = gi[2 * HDIM + j] + bih[2 * HDIM + j];
    float hr = gh[j]            + bhh[j];
    float hz = gh[HDIM + j]     + bhh[HDIM + j];
    float hn = gh[2 * HDIM + j] + bhh[2 * HDIM + j];
    float r = 1.0f / (1.0f + __expf(-(ir + hr)));
    float z = 1.0f / (1.0f + __expf(-(iz + hz)));
    float n = tanhf(in + r * hn);
    float hv = (1.0f - z) * n + z * h[i];
    h[i] = hv;
    hh[i] = (_Float16)hv;
    if (ys) ys[i] = (_Float16)hv;
}

// ---------- K parallel decoder heads ----------
__global__ void decoder(const float* __restrict__ hf, const float* __restrict__ W1,
                        const float* __restrict__ b1, const float* __restrict__ W2,
                        const float* __restrict__ b2, float* __restrict__ out) {
    int i = blockIdx.x * blockDim.x + threadIdx.x;   // BATCH*KH
    if (i >= BATCH * KH) return;
    int b = i / KH, k = i % KH;
    const float* hb = hf + (long long)b * HDIM;
    const float* w1 = W1 + (long long)k * HDIM * DHD;
    float acc = b2[k];
#pragma unroll 4
    for (int d = 0; d < DHD; ++d) {
        float s = b1[k * DHD + d];
        for (int hcol = 0; hcol < HDIM; ++hcol)
            s = fmaf(hb[hcol], w1[hcol * DHD + d], s);
        acc = fmaf(fmaxf(s, 0.0f), W2[k * DHD + d], acc);
    }
    out[i] = acc;
}

extern "C" void kernel_launch(void* const* d_in, const int* in_sizes, int n_in,
                              void* d_out, int out_size, void* d_ws, size_t ws_size,
                              hipStream_t stream) {
    const float* x        = (const float*)d_in[0];
    const float* eattr    = (const float*)d_in[1];
    const float* Wr0      = (const float*)d_in[2];
    const float* Wn0      = (const float*)d_in[3];
    const float* b0       = (const float*)d_in[4];
    const float* Wr1      = (const float*)d_in[5];
    const float* Wn1      = (const float*)d_in[6];
    const float* b1       = (const float*)d_in[7];
    const float* emptyE   = (const float*)d_in[8];
    const float* Wih0     = (const float*)d_in[9];
    const float* Whh0     = (const float*)d_in[10];
    const float* bih0     = (const float*)d_in[11];
    const float* bhh0     = (const float*)d_in[12];
    const float* Wih1     = (const float*)d_in[13];
    const float* Whh1     = (const float*)d_in[14];
    const float* bih1     = (const float*)d_in[15];
    const float* bhh1     = (const float*)d_in[16];
    const float* dW1      = (const float*)d_in[17];
    const float* db1      = (const float*)d_in[18];
    const float* dW2      = (const float*)d_in[19];
    const float* db2      = (const float*)d_in[20];
    const int*   eidx     = (const int*)d_in[21];
    const int*   labels   = (const int*)d_in[22];
    const int*   labelmap = (const int*)d_in[23];
    const int* src = eidx;
    const int* dst = eidx + N_EDGES;
    float* out = (float*)d_out;

    // ---- workspace carve-out ----
    char* p = (char*)d_ws;
    auto carve = [&](size_t bytes) { void* r = (void*)p; p += (bytes + 255) & ~(size_t)255; return r; };
    _Float16* h1buf  = (_Float16*)carve((size_t)N_NODES * FDIM * 2);
    _Float16* aggbuf = (_Float16*)carve((size_t)N_NODES * FDIM * 2);
    float*    agg0   = (float*)   carve((size_t)N_NODES * F_INP * 4);
    float*    sums   = (float*)   carve((size_t)NGR * FDIM * 4);
    int*      cnts   = (int*)     carve((size_t)NGR * 4);
    _Float16* xs     = (_Float16*)carve((size_t)TSTEPS * BATCH * FDIM * 2);
    _Float16* ys0    = (_Float16*)carve((size_t)TSTEPS * BATCH * HDIM * 2);
    float*    GI     = (float*)   carve((size_t)BATCH * G3H * 4);
    float*    GH     = (float*)   carve((size_t)BATCH * G3H * 4);
    float*    h0f    = (float*)   carve((size_t)BATCH * HDIM * 4);
    _Float16* h0h    = (_Float16*)carve((size_t)BATCH * HDIM * 2);
    float*    h1f    = (float*)   carve((size_t)BATCH * HDIM * 4);
    _Float16* h1h    = (_Float16*)carve((size_t)BATCH * HDIM * 2);
    _Float16* wih0h  = (_Float16*)carve((size_t)G3H * FDIM * 2);
    _Float16* whh0h  = (_Float16*)carve((size_t)G3H * HDIM * 2);
    _Float16* wih1h  = (_Float16*)carve((size_t)G3H * HDIM * 2);
    _Float16* whh1h  = (_Float16*)carve((size_t)G3H * HDIM * 2);
    _Float16* btc    = (_Float16*)carve((size_t)FDIM * HDIM * 2);

    // ---- zero accumulators / state (every call; graph-capture safe) ----
    hipMemsetAsync(aggbuf, 0, (size_t)N_NODES * FDIM * 2, stream);
    hipMemsetAsync(agg0,   0, (size_t)N_NODES * F_INP * 4, stream);
    hipMemsetAsync(sums,   0, (size_t)NGR * FDIM * 4, stream);
    hipMemsetAsync(cnts,   0, (size_t)NGR * 4, stream);
    hipMemsetAsync(h0f,    0, (size_t)BATCH * HDIM * 4, stream);
    hipMemsetAsync(h0h,    0, (size_t)BATCH * HDIM * 2, stream);
    hipMemsetAsync(h1f,    0, (size_t)BATCH * HDIM * 4, stream);
    hipMemsetAsync(h1h,    0, (size_t)BATCH * HDIM * 2, stream);

    // ---- weight prep (f16 / transposed) ----
    cvt_f32_f16<<<(G3H * FDIM + 255) / 256, 256, 0, stream>>>(Wih0, wih0h, G3H * FDIM);
    cvt_f32_f16<<<(G3H * HDIM + 255) / 256, 256, 0, stream>>>(Whh0, whh0h, G3H * HDIM);
    cvt_f32_f16<<<(G3H * HDIM + 255) / 256, 256, 0, stream>>>(Wih1, wih1h, G3H * HDIM);
    cvt_f32_f16<<<(G3H * HDIM + 255) / 256, 256, 0, stream>>>(Whh1, whh1h, G3H * HDIM);
    build_btc<<<(FDIM * HDIM + 255) / 256, 256, 0, stream>>>(Wr1, Wn1, btc);

    // ---- conv0 ----
    {
        long long th = (long long)N_EDGES * 16;
        edge_agg0<<<(unsigned)((th + 255) / 256), 256, 0, stream>>>(src, dst, eattr, x, agg0);
        long long tn = (long long)N_NODES * FDIM;
        conv0_node<<<(unsigned)((tn + 255) / 256), 256, 0, stream>>>(x, agg0, Wr0, Wn0, b0, h1buf);
    }
    // ---- conv1 edge aggregation (pk f16 atomics) ----
    {
        long long th = (long long)N_EDGES * 32;
        edge_agg1<<<(unsigned)((th + 255) / 256), 256, 0, stream>>>(src, dst, eattr, h1buf, aggbuf);
    }
    // ---- counts + conv1 dense (WMMA) fused with pooling ----
    count_labels<<<(N_NODES + 255) / 256, 256, 0, stream>>>(labels, cnts);
    {
        int total = N_NODES / 16;   // 31250 strip-waves
        conv1_pool<<<(total * 32 + 255) / 256, 256, 0, stream>>>(h1buf, aggbuf, btc, b1, labels, sums);
    }
    // ---- pool finalize -> xs [T,B,F] f16 ----
    fill_xs<<<(TSTEPS * BATCH * FDIM + 255) / 256, 256, 0, stream>>>(emptyE, xs);
    scatter_xs<<<(NGR * FDIM + 255) / 256, 256, 0, stream>>>(sums, cnts, labelmap, xs);

    // ---- GRU: per step, GI = xt @ WihT, GH = h @ WhhT, then gate math ----
    const int NgG = G3H / 64;                // 6 strip-groups over N=384
    const int totG = (BATCH / 16) * NgG;     // 768 waves
    const int gemmBlocks = (totG * 32 + 255) / 256;
    for (int t = 0; t < TSTEPS; ++t) {       // layer 0 (input F=64)
        gemm16x64<<<gemmBlocks, 256, 0, stream>>>(xs + (size_t)t * BATCH * FDIM, FDIM,
                                                  wih0h, FDIM, GI, G3H, NgG, totG, FDIM);
        gemm16x64<<<gemmBlocks, 256, 0, stream>>>(h0h, HDIM, whh0h, HDIM, GH, G3H, NgG, totG, HDIM);
        gru_gate<<<(BATCH * HDIM + 255) / 256, 256, 0, stream>>>(
            GI, GH, bih0, bhh0, h0f, h0h, ys0 + (size_t)t * BATCH * HDIM);
    }
    for (int t = 0; t < TSTEPS; ++t) {       // layer 1 (input H=128)
        gemm16x64<<<gemmBlocks, 256, 0, stream>>>(ys0 + (size_t)t * BATCH * HDIM, HDIM,
                                                  wih1h, HDIM, GI, G3H, NgG, totG, HDIM);
        gemm16x64<<<gemmBlocks, 256, 0, stream>>>(h1h, HDIM, whh1h, HDIM, GH, G3H, NgG, totG, HDIM);
        gru_gate<<<(BATCH * HDIM + 255) / 256, 256, 0, stream>>>(
            GI, GH, bih1, bhh1, h1f, h1h, (_Float16*)nullptr);
    }
    // ---- decoder heads ----
    decoder<<<(BATCH * KH + 255) / 256, 256, 0, stream>>>(h1f, dW1, db1, dW2, db2, out);
}
